// _WindowCrossAttention3D_80496277062475
// MI455X (gfx1250) — compile-verified, matrix operand order checked by reading, not run on hardware
//
#include <hip/hip_runtime.h>
#include <hip/hip_bf16.h>

typedef __attribute__((ext_vector_type(16))) _Float16 v16h;
typedef __attribute__((ext_vector_type(8)))  _Float16 v8h;
typedef __attribute__((ext_vector_type(8)))  float    v8f;
typedef __attribute__((ext_vector_type(4)))  unsigned int u32x4;
typedef __attribute__((ext_vector_type(8)))  int      i32x8;
typedef __attribute__((ext_vector_type(4)))  int      i32x4;

#define LDX 200   // padded row stride (halfs) for X matrices (100 dw; 36 mod 64 row step -> conflict-free)
#define LDT 72    // padded row stride (halfs) for 64-wide (token) matrices
// Weight LDS layout written by TDM with pad_interval=32dw / pad_amount=4dw:
// 64-half chunks separated by 8-half gaps -> addr(row,k) = row*216 + (k>>6)*72 + (k&63)
#define WROW 216  // effective weight row stride in halfs (108 dw; 44 mod 64 step -> conflict-free)

// ---- generic 16x32 f16 WMMA fragment load (A-style and B-style identical):
// lane L: row = row0 + L%16 ; element j: k = k0 + (j/8)*16 + (L/16)*8 + j%8
__device__ __forceinline__ v16h ld_frag(const _Float16* m, int row0, int k0, int ld) {
  const int lane = threadIdx.x & 31;
  const int ln = lane & 15, hi = lane >> 4;
  const _Float16* p = m + (row0 + ln) * ld + k0 + hi * 8;
  v8h lo = *(const v8h*)(p);
  v8h hh = *(const v8h*)(p + 16);
  v16h r;
#pragma unroll
  for (int i = 0; i < 8; ++i) { r[i] = lo[i]; r[i + 8] = hh[i]; }
  return r;
}

// ---- same, but for the TDM chunk+gap weight layout (k0 multiple of 32 keeps
// the whole 32-wide k-step inside one 64-half chunk).
__device__ __forceinline__ v16h ld_fragW(const _Float16* m, int row0, int k0) {
  const int lane = threadIdx.x & 31;
  const int ln = lane & 15, hi = lane >> 4;
  const _Float16* p = m + (row0 + ln) * WROW + ((k0 >> 6) * 72 + (k0 & 63)) + hi * 8;
  v8h lo = *(const v8h*)(p);
  v8h hh = *(const v8h*)(p + 16);
  v16h r;
#pragma unroll
  for (int i = 0; i < 8; ++i) { r[i] = lo[i]; r[i + 8] = hh[i]; }
  return r;
}

// ---- Tensor Data Mover: async copy 192x192 f16 weight matrix (contiguous in
// global) into LDS with 8-half padding every 64 halfs. Issued by wave 0 only.
__device__ __forceinline__ void tdm_w_issue(const _Float16* gsrc, _Float16* ldst,
                                            int tid, int wave) {
#if __has_builtin(__builtin_amdgcn_tensor_load_to_lds)
  if (wave == 0) {
    unsigned long long ga = (unsigned long long)(size_t)gsrc;
    u32x4 g0;
    g0.x = 1u;                                   // count=1 (valid descriptor)
    g0.y = (unsigned)(size_t)ldst;               // lds_addr (low 32 bits of generic ptr)
    g0.z = (unsigned)ga;                         // global_addr[31:0]
    g0.w = (unsigned)((ga >> 32) & 0x01ffffffu)  // global_addr[56:32]
           | 0x80000000u;                        // type=2 ("image") in bits 127:126
    // data_size=2B | pad_enable | pad_interval=32dw | pad_amount=4dw ;
    // tensor_dim0=192, tensor_dim1=192, tile_dim0=192, tile_dim1=192, stride0=192
    i32x8 g1 = {0x07110000, 0x00C00000, 0x00C00000, 0x00C00000,
                0x000000C0, 0x000000C0, 0, 0};
    i32x4 gz = {0, 0, 0, 0};
#if defined(__clang_major__) && __clang_major__ >= 23
    i32x8 g4 = {0, 0, 0, 0, 0, 0, 0, 0};
    __builtin_amdgcn_tensor_load_to_lds(g0, g1, gz, gz, g4, 0);
#else
    __builtin_amdgcn_tensor_load_to_lds(g0, g1, gz, gz, 0);
#endif
  }
#else
  // fallback: cooperative copy into the same chunked layout
  for (int i = tid; i < 192 * 192; i += 256) {
    int r = i / 192, k = i % 192;
    ldst[r * WROW + (k >> 6) * 72 + (k & 63)] = gsrc[i];
  }
#endif
}

__device__ __forceinline__ void tdm_w_wait(int wave) {
#if __has_builtin(__builtin_amdgcn_tensor_load_to_lds)
  if (wave == 0) __builtin_amdgcn_s_wait_tensorcnt(0);
#endif
}

// [64 tokens x 192 out] = X[64x192] * W^T, K=192 (6 k-steps). 8 waves:
// wave = mt (4 m-tiles) x ng (2 n-groups of 6 n-tiles). acc[6] = f32 16x16 tiles.
__device__ __forceinline__ void gemm_64x192x192(const _Float16* X, const _Float16* W,
                                                int mt, int ng, v8f acc[6]) {
#pragma unroll
  for (int ks = 0; ks < 6; ++ks) {
    v16h a = ld_frag(X, mt * 16, ks * 32, LDX);
#pragma unroll
    for (int j = 0; j < 6; ++j) {
      v16h b = ld_fragW(W, (ng * 6 + j) * 16, ks * 32);
      acc[j] = __builtin_amdgcn_wmma_f32_16x16x32_f16(false, a, false, b, (short)0,
                                                      acc[j], false, false);
    }
  }
}

// Kernel 0: one-time f32 -> f16 conversion of the four 192x192 weight matrices.
__global__ void __launch_bounds__(256)
k_cvt_w(const float* __restrict__ wq, const float* __restrict__ wk,
        const float* __restrict__ wv, const float* __restrict__ wp,
        _Float16* __restrict__ o)
{
  int i = blockIdx.x * 256 + threadIdx.x;     // 576 blocks * 256 = 147456 exactly
  const float* src; int off;
  if (i < 36864)       { src = wq; off = 0; }
  else if (i < 73728)  { src = wk; off = 36864; }
  else if (i < 110592) { src = wv; off = 73728; }
  else                 { src = wp; off = 110592; }
  o[i] = (_Float16)src[i - off];
}

// Kernel 1: fused qkv projection (windowed gather) + multi-head attention.
// One block = one 4x4x4 window (2048 blocks). Writes attn output [win][64][192] f16.
__global__ void __launch_bounds__(256)
k_qkv_attn(const float* __restrict__ qin, const float* __restrict__ kvin,
           const float* __restrict__ bq, const float* __restrict__ bk,
           const float* __restrict__ bv, const _Float16* __restrict__ w16,
           _Float16* __restrict__ aout)
{
  __shared__ __align__(16) _Float16 sQ [64 * LDX];   // Xq, then q (f16)
  __shared__ __align__(16) _Float16 sKV[64 * LDX];   // Xkv
  __shared__ __align__(16) _Float16 sK [64 * LDX];   // k (f16)
  __shared__ __align__(16) _Float16 sW [192 * WROW]; // TDM weight tile; reused: vT + attn scratch
  _Float16* vT    = sW;                 // [192][LDT]  (v transposed: row=channel, col=token)
  _Float16* attnS = sW + 192 * LDT;     // 8 waves x [16][LDT]

  const int win = blockIdx.x;
  const int b  = win >> 9;
  const int wx = (win >> 6) & 7, wy = (win >> 3) & 7, wz = win & 7;
  const int tid = threadIdx.x;
  const int wave = tid >> 5, lane = tid & 31;
  const int ln = lane & 15, hi = lane >> 4;
  const int mt = wave & 3, ng = wave >> 2;

  const _Float16* wq16 = w16;
  const _Float16* wk16 = w16 + 36864;
  const _Float16* wv16 = w16 + 73728;

  tdm_w_issue(wq16, sW, tid, wave);     // DMA Wq while we gather

  const float* qb  = qin  + (size_t)b * 192 * 32768;
  const float* kvb = kvin + (size_t)b * 192 * 32768;
  // Gather shifted window: token n = lx*16+ly*4+lz, coord = (w*4 + l + 2) mod 32
  for (int i = tid; i < 64 * 192; i += 256) {
    int n = i & 63, c = i >> 6;
    int lz = n & 3, ly = (n >> 2) & 3, lx = n >> 4;
    int x = (wx * 4 + lx + 2) & 31;
    int y = (wy * 4 + ly + 2) & 31;
    int z = (wz * 4 + lz + 2) & 31;
    int g = c * 32768 + x * 1024 + y * 32 + z;
    sQ [n * LDX + c] = (_Float16)qb[g];
    sKV[n * LDX + c] = (_Float16)kvb[g];
  }

  const v8f zf = {0.f, 0.f, 0.f, 0.f, 0.f, 0.f, 0.f, 0.f};
  v8f acc[6];

  // ---------------- Q = Xq * Wq^T + bq  -> sQ (f16, [token][c]) ----------------
  tdm_w_wait(wave);
  __syncthreads();
#pragma unroll
  for (int j = 0; j < 6; ++j) acc[j] = zf;
  gemm_64x192x192(sQ, sW, mt, ng, acc);
  __syncthreads();
  tdm_w_issue(wk16, sW, tid, wave);     // DMA Wk while we write the q epilogue
#pragma unroll
  for (int j = 0; j < 6; ++j) {
    int c = (ng * 6 + j) * 16 + ln;
    float bb = bq[c];
#pragma unroll
    for (int r = 0; r < 8; ++r)
      sQ[(mt * 16 + hi * 8 + r) * LDX + c] = (_Float16)(acc[j][r] + bb);
  }
  tdm_w_wait(wave);
  __syncthreads();
  // ---------------- K = Xkv * Wk^T + bk -> sK ----------------
#pragma unroll
  for (int j = 0; j < 6; ++j) acc[j] = zf;
  gemm_64x192x192(sKV, sW, mt, ng, acc);
  __syncthreads();
  tdm_w_issue(wv16, sW, tid, wave);     // DMA Wv while we write the k epilogue
#pragma unroll
  for (int j = 0; j < 6; ++j) {
    int c = (ng * 6 + j) * 16 + ln;
    float bb = bk[c];
#pragma unroll
    for (int r = 0; r < 8; ++r)
      sK[(mt * 16 + hi * 8 + r) * LDX + c] = (_Float16)(acc[j][r] + bb);
  }
  tdm_w_wait(wave);
  __syncthreads();
  // ---------------- V = Xkv * Wv^T + bv -> vT (transposed, over W region) ----------------
#pragma unroll
  for (int j = 0; j < 6; ++j) acc[j] = zf;
  gemm_64x192x192(sKV, sW, mt, ng, acc);
  __syncthreads();   // all Wv reads done; safe to overwrite W region with vT
#pragma unroll
  for (int j = 0; j < 6; ++j) {
    int c = (ng * 6 + j) * 16 + ln;
    float bb = bv[c];
    v8h pk;
#pragma unroll
    for (int r = 0; r < 8; ++r) pk[r] = (_Float16)(acc[j][r] + bb);
    *(v8h*)(vT + c * LDT + mt * 16 + hi * 8) = pk;   // 8 contiguous tokens -> b128
  }
  __syncthreads();

  // ---------------- Attention: 24 row-blocks = 6 heads x 4 m-tiles ----------------
  const float scl = 0.17677669529663687f;  // 1/sqrt(32)
  for (int rb = wave; rb < 24; rb += 8) {
    const int h = rb >> 2, qmt = rb & 3;
    v16h qa = ld_frag(sQ, qmt * 16, h * 32, LDX);
    v8f s[4];
#pragma unroll
    for (int nt = 0; nt < 4; ++nt) {
      v16h kb = ld_frag(sK, nt * 16, h * 32, LDX);   // B[d][n]=k[n][d]
      s[nt] = __builtin_amdgcn_wmma_f32_16x16x32_f16(false, qa, false, kb, (short)0,
                                                     zf, false, false);
    }
    // softmax: row m = r + 8*hi spans 16 lanes x 4 tiles; reduce within 16-lane half
#pragma unroll
    for (int r = 0; r < 8; ++r) {
      float a0 = s[0][r] * scl, a1 = s[1][r] * scl, a2 = s[2][r] * scl, a3 = s[3][r] * scl;
      float mx = fmaxf(fmaxf(a0, a1), fmaxf(a2, a3));
      mx = fmaxf(mx, __shfl_xor(mx, 1, 32));
      mx = fmaxf(mx, __shfl_xor(mx, 2, 32));
      mx = fmaxf(mx, __shfl_xor(mx, 4, 32));
      mx = fmaxf(mx, __shfl_xor(mx, 8, 32));
      a0 = __expf(a0 - mx); a1 = __expf(a1 - mx); a2 = __expf(a2 - mx); a3 = __expf(a3 - mx);
      float sm = a0 + a1 + a2 + a3;
      sm += __shfl_xor(sm, 1, 32);
      sm += __shfl_xor(sm, 2, 32);
      sm += __shfl_xor(sm, 4, 32);
      sm += __shfl_xor(sm, 8, 32);
      float inv = 1.0f / sm;
      s[0][r] = a0 * inv; s[1][r] = a1 * inv; s[2][r] = a2 * inv; s[3][r] = a3 * inv;
    }
    // D layout (n over lanes) -> A layout (m over lanes): transpose through wave-private LDS
    _Float16* aw = attnS + wave * 16 * LDT;
#pragma unroll
    for (int nt = 0; nt < 4; ++nt)
#pragma unroll
      for (int r = 0; r < 8; ++r)
        aw[(hi * 8 + r) * LDT + nt * 16 + ln] = (_Float16)s[nt][r];
    // out = attn[16x64] @ v_h[64x32]
#pragma unroll
    for (int dt = 0; dt < 2; ++dt) {
      v8f o = zf;
#pragma unroll
      for (int ks = 0; ks < 2; ++ks) {
        v16h a  = ld_frag(aw, 0, ks * 32, LDT);
        v16h bf = ld_frag(vT, h * 32 + dt * 16, ks * 32, LDT);  // B[token][d]=vT[d][token]
        o = __builtin_amdgcn_wmma_f32_16x16x32_f16(false, a, false, bf, (short)0,
                                                   o, false, false);
      }
      int c = h * 32 + dt * 16 + ln;
      _Float16* dst = aout + (size_t)win * 12288 + c;
#pragma unroll
      for (int r = 0; r < 8; ++r)
        dst[(qmt * 16 + hi * 8 + r) * 192] = (_Float16)o[r];
    }
  }
}

// Kernel 2: proj conv1x1 per window + un-window/roll scatter with bias, f32 output.
__global__ void __launch_bounds__(256)
k_proj(const _Float16* __restrict__ ain, const _Float16* __restrict__ wp16,
       const float* __restrict__ bp, float* __restrict__ out)
{
  __shared__ __align__(16) _Float16 sX[64 * LDX];
  __shared__ __align__(16) _Float16 sW[192 * WROW];
  float* st = (float*)sW;  // f32 staging [64][LDX] = 51.2KB <= 82.9KB (after GEMM)

  const int win = blockIdx.x;
  const int tid = threadIdx.x;
  const int wave = tid >> 5, lane = tid & 31, ln = lane & 15, hi = lane >> 4;
  const int mt = wave & 3, ng = wave >> 2;

  tdm_w_issue(wp16, sW, tid, wave);     // DMA Wp while we stage X

  const _Float16* xin = ain + (size_t)win * 12288;
  for (int i = tid; i < 1536; i += 256) {      // vectorized b128 loads (contiguous ws)
    int e = i * 8; int n = e / 192, c = e % 192;
    *(v8h*)(sX + n * LDX + c) = *(const v8h*)(xin + e);
  }
  tdm_w_wait(wave);
  __syncthreads();

  const v8f zf = {0.f, 0.f, 0.f, 0.f, 0.f, 0.f, 0.f, 0.f};
  v8f acc[6];
#pragma unroll
  for (int j = 0; j < 6; ++j) acc[j] = zf;
  gemm_64x192x192(sX, sW, mt, ng, acc);
  __syncthreads();
#pragma unroll
  for (int j = 0; j < 6; ++j) {
    int c = (ng * 6 + j) * 16 + ln;
#pragma unroll
    for (int r = 0; r < 8; ++r)
      st[(mt * 16 + hi * 8 + r) * LDX + c] = acc[j][r];
  }
  __syncthreads();

  const int b = win >> 9, wx = (win >> 6) & 7, wy = (win >> 3) & 7, wz = win & 7;
  float* ob = out + (size_t)b * 192 * 32768;
  for (int i = tid; i < 64 * 192; i += 256) {
    int n = i & 63, c = i >> 6;
    int lz = n & 3, ly = (n >> 2) & 3, lx = n >> 4;
    int x = (wx * 4 + lx + 2) & 31;
    int y = (wy * 4 + ly + 2) & 31;
    int z = (wz * 4 + lz + 2) & 31;
    ob[c * 32768 + x * 1024 + y * 32 + z] = st[n * LDX + c] + bp[c];
  }
}

extern "C" void kernel_launch(void* const* d_in, const int* in_sizes, int n_in,
                              void* d_out, int out_size, void* d_ws, size_t ws_size,
                              hipStream_t stream) {
  const float* query = (const float*)d_in[0];
  const float* keyv  = (const float*)d_in[1];
  const float* wq = (const float*)d_in[2]; const float* bq = (const float*)d_in[3];
  const float* wk = (const float*)d_in[4]; const float* bk = (const float*)d_in[5];
  const float* wv = (const float*)d_in[6]; const float* bv = (const float*)d_in[7];
  const float* wp = (const float*)d_in[8]; const float* bp = (const float*)d_in[9];
  _Float16* attn = (_Float16*)d_ws;            // 2048*64*192 f16 = 48 MiB
  _Float16* w16  = attn + (size_t)2048 * 12288; // +288 KiB of f16 weights
  float* out = (float*)d_out;
  (void)in_sizes; (void)n_in; (void)out_size; (void)ws_size;
  k_cvt_w<<<576, 256, 0, stream>>>(wq, wk, wv, wp, w16);
  k_qkv_attn<<<2048, 256, 0, stream>>>(query, keyv, bq, bk, bv, w16, attn);
  k_proj<<<2048, 256, 0, stream>>>(attn, w16 + 3 * 36864, bp, out);
}
